// DirectedGraphLearner_82806969466946
// MI455X (gfx1250) — matrix operand order repaired and checked by reading, not compiled, not run on hardware
//
#include <hip/hip_runtime.h>
#include <hip/hip_bf16.h>

typedef __bf16 v16bf __attribute__((ext_vector_type(16)));
typedef __bf16 v8bf  __attribute__((ext_vector_type(8)));
typedef float  v8f   __attribute__((ext_vector_type(8)));

#define B_   8
#define N_   1024
#define D_   256
#define H_   4
#define HD_  64
#define KTOP 153
#define LDSTR 1024

__device__ __forceinline__ float gelu_erf(float x) {
    return 0.5f * x * (1.0f + erff(x * 0.70710678118654752440f));
}

// ---------- prep: x (fp32) -> bf16, same layout [B*N][D] ----------
__global__ void cvt_x_bf16(const float* __restrict__ x, __bf16* __restrict__ xb, int n) {
    int i = blockIdx.x * blockDim.x + threadIdx.x;
    if (i < n) xb[i] = (__bf16)x[i];
}

// ---------- prep: W (D x D, row = input dim) -> Wt bf16 [out][in] for both mats ----------
__global__ void cvt_w_bf16_t(const float* __restrict__ Wsrc, const float* __restrict__ Wtgt,
                             __bf16* __restrict__ WtS, __bf16* __restrict__ WtT) {
    int tid = blockIdx.x * blockDim.x + threadIdx.x;   // 2*256*256 threads
    int mat = tid >> 16;
    int o   = (tid >> 8) & 255;
    int i   = tid & 255;
    const float* W  = mat ? Wtgt : Wsrc;
    __bf16*      Wt = mat ? WtT  : WtS;
    Wt[o * D_ + i] = (__bf16)W[i * D_ + o];
}

// ---------- projections: S = x @ Wsrc, T = x @ Wtgt, bf16 in / bf16 out, f32 accum ----------
// One wave per 16x16 output tile. 16384 tiles total (2 matrices x 512 x 16), 8 waves/block.
__global__ void proj_gemm(const __bf16* __restrict__ xb,
                          const __bf16* __restrict__ WtS, const __bf16* __restrict__ WtT,
                          __bf16* __restrict__ S, __bf16* __restrict__ T) {
    const int lane = threadIdx.x & 31;
    const int wave = threadIdx.x >> 5;
    const int tile = blockIdx.x * 8 + wave;      // 0 .. 16383
    const int mat  = tile >> 13;                 // 8192 tiles per matrix
    const int rem  = tile & 8191;
    const int tm   = rem >> 4;                   // 0 .. 511
    const int tn   = rem & 15;                   // 0 .. 15
    const __bf16* Wt  = mat ? WtT : WtS;
    __bf16*       Out = mat ? T   : S;

    const int m0   = tm * 16, n0 = tn * 16;
    const int half = lane >> 4;                  // 0: lanes 0-15, 1: lanes 16-31
    const int l15  = lane & 15;

    const __bf16* arow = xb + (m0 + l15) * D_;   // A row for this lane's M
    const __bf16* brow = Wt + (n0 + l15) * D_;   // B column (stored as Wt row)

    v8f c = {};
    #pragma unroll
    for (int kk = 0; kk < D_; kk += 32) {
        // A 16x32 bf16: lanes<16 hold K 0..15 contiguous; lanes>=16 hold K 16..31
        union { v16bf v; v8bf h2[2]; } a;
        a.v = *(const v16bf*)(arow + kk + half * 16);
        // B 32x16 bf16: lanes<16 hold K 0..15 contiguous; lanes>=16 hold K 16..31
        v16bf bm = *(const v16bf*)(brow + kk + half * 16);
        c = __builtin_amdgcn_wmma_f32_16x16x32_bf16(false, a.v, false, bm,
                                                    (short)0, c, false, false);
    }
    // D 16x16 f32: VGPR r -> M = r + half*8, N = l15
    #pragma unroll
    for (int r = 0; r < 8; ++r) {
        const int row = m0 + r + half * 8;
        Out[row * D_ + n0 + l15] = (__bf16)c[r];
    }
}

// ---------- fused: adj tile GEMM + exact-erf GELU + per-row top-k mask + store ----------
// One block per (b, h, 16-row block): 8*4*64 = 2048 blocks, 4 waves each.
__global__ void __launch_bounds__(128)
adj_topk(const __bf16* __restrict__ S, const __bf16* __restrict__ T,
         float* __restrict__ out) {
    __shared__ float lds[16 * LDSTR];            // 64 KB: 16 rows x 1024 gelu'd scores
    const int lane = threadIdx.x & 31;
    const int wave = threadIdx.x >> 5;
    const int blk  = blockIdx.x;
    const int b    = blk >> 8;                   // / (H*64)
    const int h    = (blk >> 6) & 3;
    const int i0   = (blk & 63) * 16;
    const int half = lane >> 4;
    const int l15  = lane & 15;

    // A tile: src rows i0..i0+15, K = head dims 0..63 (two k-steps of 32), loaded once
    const __bf16* srow = S + (b * N_ + i0 + l15) * D_ + h * HD_;
    union { v16bf v; v8bf h2[2]; } a0, a1;
    a0.v = *(const v16bf*)(srow + half * 16);
    a1.v = *(const v16bf*)(srow + 32 + half * 16);

    const __bf16* tbase = T + b * N_ * D_ + h * HD_;

    for (int jt = wave; jt < 64; jt += 4) {      // 16 j-tiles per wave, uniform count
        const int j0 = jt * 16;
        const __bf16* trow = tbase + (j0 + l15) * D_;   // tgt row j = column of B
        v16bf b0 = *(const v16bf*)(trow + half * 16);
        v16bf b1 = *(const v16bf*)(trow + 32 + half * 16);
        v8f c = {};
        c = __builtin_amdgcn_wmma_f32_16x16x32_bf16(false, a0.v, false, b0,
                                                    (short)0, c, false, false);
        c = __builtin_amdgcn_wmma_f32_16x16x32_bf16(false, a1.v, false, b1,
                                                    (short)0, c, false, false);
        #pragma unroll
        for (int r = 0; r < 8; ++r) {
            const int m = r + half * 8;
            lds[m * LDSTR + j0 + l15] = gelu_erf(c[r]);
        }
    }
    __syncthreads();

    // Per-row top-KTOP: each wave owns 4 rows. Lane owns 8 float4 chunks at
    // cols lane*4 + 128*t. Radix/binary search on order-preserving keys with
    // ballot+popcount counting (SALU co-issue, no cross-lane shuffles).
    for (int rr = 0; rr < 4; ++rr) {
        const int row = wave * 4 + rr;
        float    fv[32];
        unsigned uk[32];
        #pragma unroll
        for (int t = 0; t < 8; ++t) {
            const float4 q = *(const float4*)&lds[row * LDSTR + lane * 4 + 128 * t];
            fv[4 * t + 0] = q.x; fv[4 * t + 1] = q.y;
            fv[4 * t + 2] = q.z; fv[4 * t + 3] = q.w;
        }
        #pragma unroll
        for (int t = 0; t < 32; ++t) {
            unsigned u = __float_as_uint(fv[t]);
            uk[t] = (u & 0x80000000u) ? ~u : (u | 0x80000000u);  // monotonic key
        }
        unsigned thr = 0;
        for (int bit = 31; bit >= 0; --bit) {
            const unsigned cand = thr | (1u << bit);
            int cnt = 0;
            #pragma unroll
            for (int t = 0; t < 32; ++t) {
                // ballot aggregates across all 32 lanes; popcount on SALU
                cnt += __popcll(__ballot(uk[t] >= cand));
            }
            if (cnt >= KTOP) thr = cand;          // cnt uniform across wave
        }
        float* orow = out + (long)(((b * H_ + h) * N_) + i0 + row) * N_;
        #pragma unroll
        for (int t = 0; t < 8; ++t) {
            float4 o;
            o.x = (uk[4 * t + 0] >= thr) ? fv[4 * t + 0] : 0.0f;
            o.y = (uk[4 * t + 1] >= thr) ? fv[4 * t + 1] : 0.0f;
            o.z = (uk[4 * t + 2] >= thr) ? fv[4 * t + 2] : 0.0f;
            o.w = (uk[4 * t + 3] >= thr) ? fv[4 * t + 3] : 0.0f;
            *(float4*)&orow[lane * 4 + 128 * t] = o;
        }
    }
}

extern "C" void kernel_launch(void* const* d_in, const int* in_sizes, int n_in,
                              void* d_out, int out_size, void* d_ws, size_t ws_size,
                              hipStream_t stream) {
    const float* x    = (const float*)d_in[0];
    const float* Wsrc = (const float*)d_in[1];
    const float* Wtgt = (const float*)d_in[2];
    float* out = (float*)d_out;

    char* ws = (char*)d_ws;
    __bf16* xb  = (__bf16*)(ws);                                   // 8192*256 bf16 = 4 MB
    __bf16* WtS = (__bf16*)(ws + 4 * 1024 * 1024);                 // 128 KB
    __bf16* WtT = (__bf16*)(ws + 4 * 1024 * 1024 + 128 * 1024);    // 128 KB
    __bf16* S   = (__bf16*)(ws + 4 * 1024 * 1024 + 256 * 1024);    // 4 MB
    __bf16* Tm  = (__bf16*)(ws + 8 * 1024 * 1024 + 256 * 1024);    // 4 MB

    const int nx = B_ * N_ * D_;
    cvt_x_bf16<<<(nx + 255) / 256, 256, 0, stream>>>(x, xb, nx);
    cvt_w_bf16_t<<<(2 * D_ * D_) / 256, 256, 0, stream>>>(Wsrc, Wtgt, WtS, WtT);
    proj_gemm<<<2048, 256, 0, stream>>>(xb, WtS, WtT, S, Tm);
    adj_topk<<<2048, 128, 0, stream>>>(S, Tm, out);
}
